// MultiHeadSparseAttention_13778255085636
// MI455X (gfx1250) — matrix operand order repaired and checked by reading, not compiled, not run on hardware
//
#include <hip/hip_runtime.h>
#include <hip/hip_bf16.h>
#include <math.h>

// ---------------------------------------------------------------------------
// MI455X (gfx1250) multi-head sparse attention, f16 WMMA pipeline.
// ---------------------------------------------------------------------------

typedef __attribute__((ext_vector_type(16))) _Float16 v16h;
typedef __attribute__((ext_vector_type(8)))  _Float16 v8h;
typedef __attribute__((ext_vector_type(8)))  float    v8f;

#define B_    2
#define S_    4096
#define D_    512
#define H_    8
#define HD_   64
#define DK_   256
#define NCPAD 512   // padded compacted-column count (union of 2x256 top-k <= 512)

static __device__ __forceinline__ v8f vzero8f() {
  v8f z;
#pragma unroll
  for (int i = 0; i < 8; ++i) z[i] = 0.0f;
  return z;
}

static __device__ __forceinline__ v16h cat16(v8h lo, v8h hi) {
  return __builtin_shufflevector(lo, hi, 0,1,2,3,4,5,6,7,8,9,10,11,12,13,14,15);
}

// --------------------------- small utility kernels -------------------------

__global__ void cvt_f16_kernel(const float* __restrict__ src,
                               _Float16* __restrict__ dst, int n) {
  int i = blockIdx.x * 256 + threadIdx.x;
  if (i < n) dst[i] = (_Float16)src[i];
}

// Wt[n][k] = W[k][n], converted to f16. Row length of Wt is K.
__global__ void transpose_cvt_kernel(const float* __restrict__ W,
                                     _Float16* __restrict__ Wt, int K, int N) {
  int idx = blockIdx.x * 256 + threadIdx.x;
  if (idx >= K * N) return;
  int n = idx / K;
  int k = idx - n * K;
  Wt[idx] = (_Float16)W[(size_t)k * N + n];
}

__global__ void zero_i32_kernel(int* __restrict__ p, int n) {
  int i = blockIdx.x * 256 + threadIdx.x;
  if (i < n) p[i] = 0;
}

// --------------------------- WMMA GEMM (f16 in, f32 acc) -------------------
// C[m][n] = sum_k A[m][k] * Bt[n][k] + bias[n]   (Bt is W pre-transposed)
// Block: 128 threads = 4 waves. Block tile M=64, N=128; wave tile 64x32.
template <bool OUT_F32, bool GELU>
__global__ __launch_bounds__(128) void gemm_f16_kernel(
    const _Float16* __restrict__ A, const _Float16* __restrict__ Bt,
    const float* __restrict__ bias, void* __restrict__ Cout,
    int M, int N, int K, int ldc) {
  const int lane = threadIdx.x & 31;
  const int wave = threadIdx.x >> 5;
  const int half = lane >> 4;   // K-half selector (wave32 fragment layout)
  const int l16  = lane & 15;
  const int bm = blockIdx.y * 64;
  const int bn = blockIdx.x * 128 + wave * 32;

  v8f acc[4][2];
#pragma unroll
  for (int s = 0; s < 4; ++s)
#pragma unroll
    for (int t = 0; t < 2; ++t) acc[s][t] = vzero8f();

  for (int k0 = 0; k0 < K; k0 += 32) {
    if (k0 + 32 < K)  // gfx1250 global_prefetch_b8 of next A k-slab
      __builtin_prefetch(A + (size_t)(bm + l16) * K + k0 + 32, 0, 1);
    v16h a[4];
#pragma unroll
    for (int s = 0; s < 4; ++s) {
      // A 16x32 frag: lane m=l16; VGPR0-3 <- k0+half*8..+7, VGPR4-7 <- +16
      const _Float16* p = A + (size_t)(bm + s * 16 + l16) * K + k0 + half * 8;
      a[s] = cat16(*(const v8h*)p, *(const v8h*)(p + 16));
    }
    v16h bb[2];
#pragma unroll
    for (int t = 0; t < 2; ++t) {
      // B 32x16 frag: lane n=l16; VGPR0-3 <- k0+half*8..+7, VGPR4-7 <- +16
      const _Float16* p = Bt + (size_t)(bn + t * 16 + l16) * K + k0 + half * 8;
      bb[t] = cat16(*(const v8h*)p, *(const v8h*)(p + 16));
    }
#pragma unroll
    for (int s = 0; s < 4; ++s)
#pragma unroll
      for (int t = 0; t < 2; ++t)
        acc[s][t] = __builtin_amdgcn_wmma_f32_16x16x32_f16(
            false, a[s], false, bb[t], (short)0, acc[s][t], false, false);
  }

  // D layout: element (m,n) at lane (m>=8 ? 16:0)+n, VGPR m%8
#pragma unroll
  for (int s = 0; s < 4; ++s)
#pragma unroll
    for (int t = 0; t < 2; ++t)
#pragma unroll
      for (int r = 0; r < 8; ++r) {
        int row = bm + s * 16 + half * 8 + r;
        int col = bn + t * 16 + l16;
        float v = acc[s][t][r] + bias[col];
        if (GELU) v = 0.5f * v * (1.0f + erff(v * 0.70710678118654752f));
        if (OUT_F32)
          ((float*)Cout)[(size_t)row * ldc + col] = v;
        else
          ((_Float16*)Cout)[(size_t)row * ldc + col] = (_Float16)v;
      }
}

// --------------------------- importance scorer tail ------------------------
// imp[token] = dot(Hs[token, :256], Ws2) + bs2 ; one wave per token.
__global__ __launch_bounds__(256) void imp_kernel(
    const _Float16* __restrict__ Hs, const float* __restrict__ Ws2,
    const float* __restrict__ bs2, float* __restrict__ imp) {
  const int lane = threadIdx.x & 31;
  const int wave = threadIdx.x >> 5;
  const int token = blockIdx.x * 8 + wave;
  float s = 0.0f;
#pragma unroll
  for (int i = 0; i < DK_ / 32; ++i) {
    int kx = lane + i * 32;
    s += (float)Hs[(size_t)token * DK_ + kx] * Ws2[kx];
  }
#pragma unroll
  for (int off = 16; off > 0; off >>= 1) s += __shfl_xor(s, off, 32);
  if (lane == 0) imp[token] = s + bs2[0];
}

// --------------------------- top-k (bitonic, per batch) --------------------
__global__ __launch_bounds__(1024) void topk_kernel(
    const float* __restrict__ imp, const int* __restrict__ topk_p,
    int* __restrict__ mask) {
  __shared__ float v[S_];
  __shared__ int   id[S_];
  const int tid = threadIdx.x;
  const int b = blockIdx.x;
  for (int i = tid; i < S_; i += 1024) { v[i] = imp[b * S_ + i]; id[i] = i; }
  __syncthreads();
  for (int k = 2; k <= S_; k <<= 1) {
    for (int j = k >> 1; j > 0; j >>= 1) {
      for (int i = tid; i < S_; i += 1024) {
        int ixj = i ^ j;
        if (ixj > i) {
          bool bigFirst = ((i & k) == 0);  // descending sort overall
          float vi = v[i], vj = v[ixj];
          bool sw = bigFirst ? (vi < vj) : (vi > vj);
          if (sw) {
            v[i] = vj; v[ixj] = vi;
            int t = id[i]; id[i] = id[ixj]; id[ixj] = t;
          }
        }
      }
      __syncthreads();
    }
  }
  int tk = *topk_p;
  if (tid < tk) atomicOr(&mask[id[tid]], 1);
}

// --------------------------- mask -> compact column list -------------------
__global__ void compact_kernel(const int* __restrict__ mask,
                               int* __restrict__ cols, int* __restrict__ ncols) {
  if (threadIdx.x == 0 && blockIdx.x == 0) {
    int n = 0;
    for (int i = 0; i < S_; ++i)
      if (mask[i]) cols[n++] = i;
    *ncols = n;
  }
}

// --------------------------- gather compacted K / V ------------------------
// Kc[bh][j][d] row-major (contiguous head-dim); Vt[bh][d][j] transposed so the
// P@V B-fragments read contiguously along j. Padding (j>=nc) written as zero.
__global__ __launch_bounds__(256) void gather_kernel(
    const _Float16* __restrict__ Kh, const _Float16* __restrict__ Vh,
    const int* __restrict__ cols, const int* __restrict__ ncols_p,
    _Float16* __restrict__ Kc, _Float16* __restrict__ Vt) {
  int idx = blockIdx.x * 256 + threadIdx.x;  // 16 * 512 * 64 threads
  int d  = idx & (HD_ - 1);
  int j  = (idx >> 6) & (NCPAD - 1);
  int bh = idx >> 15;
  if (bh >= B_ * H_) return;
  int b = bh >> 3, h = bh & 7;
  int nc = *ncols_p;
  _Float16 kv = (_Float16)0.0f, vv = (_Float16)0.0f;
  if (j < nc) {
    int s = cols[j];
    size_t src = ((size_t)(b * S_ + s)) * D_ + h * HD_ + d;
    kv = Kh[src];
    vv = Vh[src];
  }
  Kc[((size_t)bh * NCPAD + j) * HD_ + d] = kv;
  Vt[((size_t)bh * HD_ + d) * NCPAD + j] = vv;
}

// --------------------------- sparse attention core -------------------------
// 4 waves/WG; wave owns 16 queries. Scores -> 64KB LDS (f16), shfl softmax,
// in-place P, then P @ Vt with f32-accum WMMA; 1/rowsum folded in epilogue.
__global__ __launch_bounds__(128) void attn_kernel(
    const _Float16* __restrict__ Qh, const _Float16* __restrict__ Kc,
    const _Float16* __restrict__ Vt, const int* __restrict__ ncols_p,
    _Float16* __restrict__ Ctx) {
  __shared__ _Float16 Sbuf[4][16][NCPAD];  // exactly 64 KB
  const int lane = threadIdx.x & 31;
  const int wave = threadIdx.x >> 5;
  const int half = lane >> 4;
  const int l16  = lane & 15;
  const int qt = blockIdx.x, h = blockIdx.y, b = blockIdx.z;
  const int bh = b * H_ + h;
  const int baseQ = qt * 64 + wave * 16;
  const int nc = *ncols_p;
  const float scale = 0.125f;  // HD^-0.5

  // Q fragments for K-dim 64 (two 16x32 A-frags), rows contiguous in f16.
  const _Float16* qrow = Qh + ((size_t)(b * S_ + baseQ + l16)) * D_ + h * HD_;
  v16h aq0 = cat16(*(const v8h*)(qrow + half * 8),
                   *(const v8h*)(qrow + 16 + half * 8));
  v16h aq1 = cat16(*(const v8h*)(qrow + 32 + half * 8),
                   *(const v8h*)(qrow + 48 + half * 8));

  // Phase 1: scores = (Q . Kc^T) * scale  -> LDS f16
  const _Float16* kcb = Kc + (size_t)bh * NCPAD * HD_;
#pragma unroll 4
  for (int nt = 0; nt < NCPAD / 16; ++nt) {
    const _Float16* kp = kcb + (size_t)(nt * 16 + l16) * HD_;
    v16h b0 = cat16(*(const v8h*)(kp + half * 8),
                    *(const v8h*)(kp + 16 + half * 8));
    v16h b1 = cat16(*(const v8h*)(kp + 32 + half * 8),
                    *(const v8h*)(kp + 48 + half * 8));
    v8f c = vzero8f();
    c = __builtin_amdgcn_wmma_f32_16x16x32_f16(false, aq0, false, b0,
                                               (short)0, c, false, false);
    c = __builtin_amdgcn_wmma_f32_16x16x32_f16(false, aq1, false, b1,
                                               (short)0, c, false, false);
#pragma unroll
    for (int r = 0; r < 8; ++r)
      Sbuf[wave][half * 8 + r][nt * 16 + l16] = (_Float16)(c[r] * scale);
  }
  __syncthreads();

  // Phase 2: masked softmax. Lane handles row l16, half the columns.
  _Float16* srow = &Sbuf[wave][l16][0];
  const int jb = half * 256;
  float mx = -3.0e38f;
  for (int jj = 0; jj < 256; jj += 8) {
    v8h pv = *(const v8h*)(srow + jb + jj);
#pragma unroll
    for (int e = 0; e < 8; ++e) {
      int j = jb + jj + e;
      float f = (float)pv[e];
      if (j < nc) mx = fmaxf(mx, f);
    }
  }
  mx = fmaxf(mx, __shfl_xor(mx, 16, 32));
  float sum = 0.0f;
  for (int jj = 0; jj < 256; jj += 8) {
    v8h pv = *(const v8h*)(srow + jb + jj);
    v8h ov;
#pragma unroll
    for (int e = 0; e < 8; ++e) {
      int j = jb + jj + e;
      float p = (j < nc) ? __expf((float)pv[e] - mx) : 0.0f;
      sum += p;
      ov[e] = (_Float16)p;
    }
    *(v8h*)(srow + jb + jj) = ov;  // in-place unnormalized P
  }
  sum += __shfl_xor(sum, 16, 32);
  float inv = 1.0f / sum;  // per lane: 1/rowsum of row l16
  __syncthreads();

  // Phase 3: out(16x64) = P(16x512) @ Vc ; B-frags from transposed Vt.
  const _Float16* vtb = Vt + (size_t)bh * HD_ * NCPAD;
  v8f acc[4];
#pragma unroll
  for (int t = 0; t < 4; ++t) acc[t] = vzero8f();
#pragma unroll 2
  for (int kt = 0; kt < NCPAD / 32; ++kt) {
    const int k0 = kt * 32;
    const _Float16* ap = &Sbuf[wave][l16][k0];
    v16h a = cat16(*(const v8h*)(ap + half * 8),
                   *(const v8h*)(ap + 16 + half * 8));
#pragma unroll
    for (int t = 0; t < 4; ++t) {
      const _Float16* vp =
          vtb + (size_t)(t * 16 + l16) * NCPAD + k0 + half * 8;
      v16h bf = cat16(*(const v8h*)vp, *(const v8h*)(vp + 16));
      acc[t] = __builtin_amdgcn_wmma_f32_16x16x32_f16(false, a, false, bf,
                                                      (short)0, acc[t],
                                                      false, false);
    }
  }
#pragma unroll
  for (int t = 0; t < 4; ++t)
#pragma unroll
    for (int r = 0; r < 8; ++r) {
      float iv = __shfl(inv, half * 8 + r, 32);  // rowsum of row half*8+r
      int row = baseQ + half * 8 + r;
      int col = h * HD_ + t * 16 + l16;
      Ctx[((size_t)(b * S_ + row)) * D_ + col] = (_Float16)(acc[t][r] * iv);
    }
}

// --------------------------- launcher --------------------------------------
extern "C" void kernel_launch(void* const* d_in, const int* in_sizes, int n_in,
                              void* d_out, int out_size, void* d_ws,
                              size_t ws_size, hipStream_t stream) {
  (void)in_sizes; (void)n_in; (void)out_size; (void)ws_size;
  const float* x   = (const float*)d_in[0];
  const float* Wq  = (const float*)d_in[1];
  const float* bq  = (const float*)d_in[2];
  const float* Wk  = (const float*)d_in[3];
  const float* bk  = (const float*)d_in[4];
  const float* Wv  = (const float*)d_in[5];
  const float* bv  = (const float*)d_in[6];
  const float* Wo  = (const float*)d_in[7];
  const float* bo  = (const float*)d_in[8];
  const float* Ws1 = (const float*)d_in[9];
  const float* bs1 = (const float*)d_in[10];
  const float* Ws2 = (const float*)d_in[11];
  const float* bs2 = (const float*)d_in[12];
  const int*   tkp = (const int*)d_in[13];

  char* ws = (char*)d_ws;
  size_t off = 0;
  auto take = [&](size_t bytes) -> char* {
    char* p = ws + off;
    off = (off + bytes + 255) & ~(size_t)255;
    return p;
  };
  const size_t tokens = (size_t)B_ * S_;
  _Float16* xh   = (_Float16*)take(tokens * D_ * 2);
  _Float16* Wqt  = (_Float16*)take((size_t)D_ * D_ * 2);
  _Float16* Wkt  = (_Float16*)take((size_t)D_ * D_ * 2);
  _Float16* Wvt  = (_Float16*)take((size_t)D_ * D_ * 2);
  _Float16* Wot  = (_Float16*)take((size_t)D_ * D_ * 2);
  _Float16* Ws1t = (_Float16*)take((size_t)DK_ * D_ * 2);
  _Float16* Qh   = (_Float16*)take(tokens * D_ * 2);
  _Float16* Kh   = (_Float16*)take(tokens * D_ * 2);
  _Float16* Vh   = (_Float16*)take(tokens * D_ * 2);
  _Float16* Ctx  = (_Float16*)take(tokens * D_ * 2);
  _Float16* Hs   = (_Float16*)take(tokens * DK_ * 2);
  float*    imp  = (float*)take(tokens * 4);
  int*      mask = (int*)take((size_t)S_ * 4);
  int*      cols = (int*)take((size_t)NCPAD * 4);
  int*      ncol = (int*)take(256);
  _Float16* Kc   = (_Float16*)take((size_t)B_ * H_ * NCPAD * HD_ * 2);
  _Float16* Vt   = (_Float16*)take((size_t)B_ * H_ * HD_ * NCPAD * 2);

  // 1) precision conversion + weight transposes
  cvt_f16_kernel<<<(tokens * D_) / 256, 256, 0, stream>>>(x, xh,
                                                          (int)(tokens * D_));
  transpose_cvt_kernel<<<(D_ * D_) / 256, 256, 0, stream>>>(Wq, Wqt, D_, D_);
  transpose_cvt_kernel<<<(D_ * D_) / 256, 256, 0, stream>>>(Wk, Wkt, D_, D_);
  transpose_cvt_kernel<<<(D_ * D_) / 256, 256, 0, stream>>>(Wv, Wvt, D_, D_);
  transpose_cvt_kernel<<<(D_ * D_) / 256, 256, 0, stream>>>(Wo, Wot, D_, D_);
  transpose_cvt_kernel<<<(D_ * DK_) / 256, 256, 0, stream>>>(Ws1, Ws1t, D_, DK_);

  // 2) Q/K/V projections (WMMA)
  dim3 gproj(D_ / 128, tokens / 64);
  gemm_f16_kernel<false, false><<<gproj, 128, 0, stream>>>(xh, Wqt, bq, Qh,
      (int)tokens, D_, D_, D_);
  gemm_f16_kernel<false, false><<<gproj, 128, 0, stream>>>(xh, Wkt, bk, Kh,
      (int)tokens, D_, D_, D_);
  gemm_f16_kernel<false, false><<<gproj, 128, 0, stream>>>(xh, Wvt, bv, Vh,
      (int)tokens, D_, D_, D_);

  // 3) scorer: gelu(x @ Ws1 + bs1) then dot with Ws2
  dim3 gscr(DK_ / 128, tokens / 64);
  gemm_f16_kernel<false, true><<<gscr, 128, 0, stream>>>(xh, Ws1t, bs1, Hs,
      (int)tokens, DK_, D_, DK_);
  imp_kernel<<<tokens / 8, 256, 0, stream>>>(Hs, Ws2, bs2, imp);

  // 4) per-batch top-k -> union column mask -> compact column list
  zero_i32_kernel<<<S_ / 256, 256, 0, stream>>>(mask, S_);
  topk_kernel<<<B_, 1024, 0, stream>>>(imp, tkp, mask);
  compact_kernel<<<1, 1, 0, stream>>>(mask, cols, ncol);

  // 5) gather compacted K (row-major) and V (transposed), zero-padded
  gather_kernel<<<(B_ * H_ * NCPAD * HD_) / 256, 256, 0, stream>>>(
      Kh, Vh, cols, ncol, Kc, Vt);

  // 6) sparse attention (WMMA scores + WMMA P@V)
  dim3 gattn(S_ / 64, H_, B_);
  attn_kernel<<<gattn, 128, 0, stream>>>(Qh, Kc, Vt, ncol, Ctx);

  // 7) output projection, f32 result
  gemm_f16_kernel<true, false><<<gproj, 128, 0, stream>>>(Ctx, Wot, bo,
      (float*)d_out, (int)tokens, D_, D_, D_);
}
